// SparseMoE_Self_Attention_5111011083043
// MI455X (gfx1250) — matrix operand order repaired and checked by reading, not compiled, not run on hardware
//
#include <hip/hip_runtime.h>
#include <hip/hip_bf16.h>
#include <stdint.h>

typedef unsigned short u16;
typedef unsigned int   u32;
typedef __attribute__((ext_vector_type(16))) __bf16 v16bf;
typedef __attribute__((ext_vector_type(8)))  float  v8f;
typedef __attribute__((ext_vector_type(8)))  u16    v8us;

union BF16x16 { v16bf v; v8us h[2]; u16 s[16]; u32 u[8]; };
union XU      { v8us h; u32 u[4]; };

__device__ __forceinline__ float bf2f(u16 h) {
  union { u32 u; float f; } c; c.u = ((u32)h) << 16; return c.f;
}
__device__ __forceinline__ u16 f2bf(float f) {
  union { float f; u32 u; } c; c.f = f;
  u32 r = c.u + 0x7FFFu + ((c.u >> 16) & 1u);
  return (u16)(r >> 16);
}
__device__ __forceinline__ v8f wmma_bf16(v16bf a, v16bf b, v8f c) {
  return __builtin_amdgcn_wmma_f32_16x16x32_bf16(false, a, false, b, (short)0, c, false, false);
}
__device__ __forceinline__ v8f vzero8() {
  v8f z;
#pragma unroll
  for (int i = 0; i < 8; ++i) z[i] = 0.f;
  return z;
}
// packed bf16 multiply: d = {a.hi*b.hi, a.lo*b.lo}
__device__ __forceinline__ u32 pk_mul_bf16(u32 a, u32 b) {
  u32 d;
  asm("v_pk_mul_bf16 %0, %1, %2" : "=v"(d) : "v"(a), "v"(b));
  return d;
}
// packed f32->bf16 convert: d = {bf16(hi), bf16(lo)}
__device__ __forceinline__ u32 cvt_pk_bf16(float lo, float hi) {
  u32 d;
  asm("v_cvt_pk_bf16_f32 %0, %1, %2" : "=v"(d) : "v"(lo), "v"(hi));
  return d;
}

#define DIM   512
#define QKVD  1536
#define NEXP  8

// ---------------------------------------------------------------------------
// Kernel A: convert Wqkv (8x1536x512) and Wp (512x512) fp32 -> bf16
// ---------------------------------------------------------------------------
__global__ void cvt_weights_kernel(const float* __restrict__ wqkv,
                                   const float* __restrict__ wp,
                                   u16* __restrict__ wqb, u16* __restrict__ wpb,
                                   int n1, int n2) {
  const int total = n1 + n2;
  for (int i = blockIdx.x * blockDim.x + threadIdx.x; i < total;
       i += gridDim.x * blockDim.x) {
    if (i < n1) wqb[i] = f2bf(wqkv[i]);
    else        wpb[i - n1] = f2bf(wp[i - n1]);
  }
}

// ---------------------------------------------------------------------------
// Kernel B: gate logits + softmax + top-2 combine weights, and x -> bf16.
// One wave per token (32 lanes x 16 elems = 512). 8 tokens per 256-thr block.
// ---------------------------------------------------------------------------
__global__ void __launch_bounds__(256) gate_convert_kernel(
    const float* __restrict__ x, const float* __restrict__ Wg,
    const float* __restrict__ bg, const float* __restrict__ ebias,
    u16* __restrict__ xbf, float* __restrict__ comb) {
  const int tid  = threadIdx.x;
  const int wv   = tid >> 5, lane = tid & 31;
  const int t    = blockIdx.x * 8 + wv;
  const float* xr = x + (size_t)t * DIM;
  const int base = lane * 16;

  float s[NEXP];
#pragma unroll
  for (int e = 0; e < NEXP; ++e) s[e] = 0.f;
  u16 tb[16];
#pragma unroll
  for (int c = 0; c < 4; ++c) {
    const float4 v = *(const float4*)&xr[base + c * 4];
    const float vv[4] = {v.x, v.y, v.z, v.w};
#pragma unroll
    for (int j = 0; j < 4; ++j) {
      tb[c * 4 + j] = f2bf(vv[j]);
#pragma unroll
      for (int e = 0; e < NEXP; ++e)
        s[e] += vv[j] * Wg[e * DIM + base + c * 4 + j];
    }
  }
  v8us p0, p1;
#pragma unroll
  for (int j = 0; j < 8; ++j) { p0[j] = tb[j]; p1[j] = tb[8 + j]; }
  *(v8us*)(xbf + (size_t)t * DIM + base)     = p0;
  *(v8us*)(xbf + (size_t)t * DIM + base + 8) = p1;

#pragma unroll
  for (int e = 0; e < NEXP; ++e) {
#pragma unroll
    for (int off = 16; off > 0; off >>= 1) s[e] += __shfl_xor(s[e], off, 32);
    s[e] += bg[e];
  }
  float mx = s[0];
#pragma unroll
  for (int e = 1; e < NEXP; ++e) mx = fmaxf(mx, s[e]);
  float g[NEXP], sum = 0.f;
#pragma unroll
  for (int e = 0; e < NEXP; ++e) { g[e] = __expf(s[e] - mx); sum += g[e]; }
  const float inv = 1.f / sum;
#pragma unroll
  for (int e = 0; e < NEXP; ++e) g[e] = g[e] * inv + ebias[e];
  int i1 = 0; float v1 = g[0];
#pragma unroll
  for (int e = 1; e < NEXP; ++e) if (g[e] > v1) { v1 = g[e]; i1 = e; }
  int i2 = -1; float v2 = -1e30f;
#pragma unroll
  for (int e = 0; e < NEXP; ++e) if (e != i1 && g[e] > v2) { v2 = g[e]; i2 = e; }
  if (lane == 0) {
    float* cr = comb + (size_t)t * NEXP;
#pragma unroll
    for (int e = 0; e < NEXP; ++e)
      cr[e] = (e == i1) ? v1 : ((e == i2) ? v2 : 0.f);
  }
}

// ---------------------------------------------------------------------------
// Kernel C: fused  qkv = sum_e (c_e ⊙ x_tile) @ Wqkv_e^T  ->  8x8 head attn
//           -> out = o @ Wp^T + bp.  32 tokens/block, 512 threads (16 waves).
// LDS: [0,16384) u16 : x tile (32x512 bf16), reused as o tile after attn
//      [16384, 65536) : qkv tile (32x1536 bf16)
// ---------------------------------------------------------------------------
__global__ void __launch_bounds__(512) moe_attn_proj_kernel(
    const u16* __restrict__ xbf, const u16* __restrict__ wqb,
    const u16* __restrict__ wpb, const float* __restrict__ comb,
    const float* __restrict__ bp, float* __restrict__ out) {
  __shared__ u16 sm[16384 + 32 * QKVD];

  const int tid = threadIdx.x;
  const int wv  = tid >> 5, lane = tid & 31;
  const int hi  = lane >> 4, l15 = lane & 15;
  const int t0  = blockIdx.x * 32;

  __builtin_prefetch(wqb + (size_t)tid * 64, 0, 0);

  // -- async-copy x tile (32x512 bf16 = 32KB) from global into LDS ----------
#pragma unroll
  for (int it = 0; it < 4; ++it) {
    const int idx = (tid + it * 512) * 8;         // u16 index, 16B aligned
    const unsigned lds = (unsigned)(idx * 2);     // sm is the only LDS object -> offset 0
    const u16* gp = xbf + (size_t)t0 * DIM + idx;
    asm volatile("global_load_async_to_lds_b128 %0, %1, off"
                 :: "v"(lds), "v"(gp) : "memory");
  }

  // per-row combine weights, packed as replicated bf16 pairs for v_pk_mul_bf16
  u32 cpk[2][NEXP];
#pragma unroll
  for (int mt = 0; mt < 2; ++mt) {
    const float* cr = comb + (size_t)(t0 + mt * 16 + l15) * NEXP;
#pragma unroll
    for (int e = 0; e < NEXP; ++e) {
      const u16 cb = f2bf(cr[e]);
      cpk[mt][e] = ((u32)cb << 16) | (u32)cb;
    }
  }

  asm volatile("s_wait_asynccnt 0" ::: "memory");
  __syncthreads();

  // ---- Stage 1: qkv tile, expert-folded dense GEMM ------------------------
  // wave wv owns 96 output cols (6 n-tiles) across both 16-row m-tiles.
  const int nb = wv * 96;
  v8f acc[2][6];
#pragma unroll
  for (int mt = 0; mt < 2; ++mt)
#pragma unroll
    for (int n = 0; n < 6; ++n) acc[mt][n] = vzero8();

  for (int kk = 0; kk < 16; ++kk) {            // K = 512 = 16 * 32
    XU xr0[2], xr1[2];
#pragma unroll
    for (int mt = 0; mt < 2; ++mt) {
      const int rb = (mt * 16 + l15) * DIM + kk * 32 + hi * 8;
      xr0[mt].h = *(const v8us*)&sm[rb];
      xr1[mt].h = *(const v8us*)&sm[rb + 16];
    }
#pragma unroll
    for (int e = 0; e < NEXP; ++e) {
      BF16x16 a[2];
#pragma unroll
      for (int mt = 0; mt < 2; ++mt) {
        const u32 cp = cpk[mt][e];
#pragma unroll
        for (int i = 0; i < 4; ++i) {
          a[mt].u[i]     = pk_mul_bf16(xr0[mt].u[i], cp);
          a[mt].u[4 + i] = pk_mul_bf16(xr1[mt].u[i], cp);
        }
      }
      const u16* wqe = wqb + (size_t)e * (QKVD * DIM) + kk * 32 + hi * 8;
#pragma unroll
      for (int n = 0; n < 6; ++n) {
        const u16* p = wqe + (size_t)(nb + n * 16 + l15) * DIM;
        BF16x16 b;
        b.h[0] = *(const v8us*)p;
        b.h[1] = *(const v8us*)(p + 16);
        acc[0][n] = wmma_bf16(a[0].v, b.v, acc[0][n]);
        acc[1][n] = wmma_bf16(a[1].v, b.v, acc[1][n]);
      }
    }
  }
  // spill qkv accumulators to LDS as bf16 (C layout: m = vgpr + 8*hi, n = l15)
#pragma unroll
  for (int mt = 0; mt < 2; ++mt)
#pragma unroll
    for (int n = 0; n < 6; ++n) {
      const int col = nb + n * 16 + l15;
#pragma unroll
      for (int j = 0; j < 4; ++j) {
        const u32 pk = cvt_pk_bf16(acc[mt][n][2 * j], acc[mt][n][2 * j + 1]);
        const int m0 = mt * 16 + 2 * j + hi * 8;
        sm[16384 + m0 * QKVD + col]       = (u16)(pk & 0xFFFFu);
        sm[16384 + (m0 + 1) * QKVD + col] = (u16)(pk >> 16);
      }
    }
  __syncthreads();

  // ---- Stage 2: per-token 8x8 cross-head attention ------------------------
  if (lane < 16) {
    const int tok = 2 * wv + (lane >> 3);
    const int h   = lane & 7;
    const u16* qk = &sm[16384 + tok * QKVD];
    float s[8];
#pragma unroll
    for (int g = 0; g < 8; ++g) {
      float d = 0.f;
      for (int dd = 0; dd < 64; ++dd)
        d += bf2f(qk[h * 64 + dd]) * bf2f(qk[512 + g * 64 + dd]);
      s[g] = d * 0.125f;                        // hd^-0.5
    }
    float mx = s[0];
#pragma unroll
    for (int g = 1; g < 8; ++g) mx = fmaxf(mx, s[g]);
    float aw[8], sum = 0.f;
#pragma unroll
    for (int g = 0; g < 8; ++g) { aw[g] = __expf(s[g] - mx); sum += aw[g]; }
    const float inv = 1.f / sum;
#pragma unroll
    for (int g = 0; g < 8; ++g) aw[g] *= inv;
    for (int dd = 0; dd < 64; ++dd) {
      float o = 0.f;
#pragma unroll
      for (int g = 0; g < 8; ++g) o += aw[g] * bf2f(qk[1024 + g * 64 + dd]);
      sm[tok * DIM + dd * 8 + h] = f2bf(o);     // transposed (hd,H) layout
    }
  }
  __syncthreads();

  // ---- Stage 3: out = o @ Wp^T + bp ---------------------------------------
  const int mt3 = wv >> 3;                      // which 16-row m-tile
  const int nb3 = (wv & 7) * 64;                // 4 n-tiles per wave
  v8f a3[4];
#pragma unroll
  for (int n = 0; n < 4; ++n) a3[n] = vzero8();
  for (int kk = 0; kk < 16; ++kk) {
    const int rb = (mt3 * 16 + l15) * DIM + kk * 32 + hi * 8;
    BF16x16 a;
    a.h[0] = *(const v8us*)&sm[rb];
    a.h[1] = *(const v8us*)&sm[rb + 16];
    const u16* wpk = wpb + kk * 32 + hi * 8;
#pragma unroll
    for (int n = 0; n < 4; ++n) {
      const u16* p = wpk + (size_t)(nb3 + n * 16 + l15) * DIM;
      BF16x16 b;
      b.h[0] = *(const v8us*)p;
      b.h[1] = *(const v8us*)(p + 16);
      a3[n] = wmma_bf16(a.v, b.v, a3[n]);
    }
  }
#pragma unroll
  for (int n = 0; n < 4; ++n) {
    const int col = nb3 + n * 16 + l15;
    const float bpv = bp[col];
#pragma unroll
    for (int j = 0; j < 8; ++j) {
      const int row = t0 + mt3 * 16 + j + hi * 8;
      out[(size_t)row * DIM + col] = a3[n][j] + bpv;
    }
  }
}

// ---------------------------------------------------------------------------
extern "C" void kernel_launch(void* const* d_in, const int* in_sizes, int n_in,
                              void* d_out, int out_size, void* d_ws, size_t ws_size,
                              hipStream_t stream) {
  const float* x     = (const float*)d_in[0];
  const float* Wqkv  = (const float*)d_in[1];
  const float* Wg    = (const float*)d_in[2];
  const float* bg    = (const float*)d_in[3];
  const float* ebias = (const float*)d_in[4];
  const float* Wp    = (const float*)d_in[5];
  const float* bp    = (const float*)d_in[6];
  float* out = (float*)d_out;

  char* ws = (char*)d_ws;
  u16*   xbf  = (u16*)(ws);                                    // 64 MB
  u16*   wqb  = (u16*)(ws + 67108864);                         // 12 MB
  u16*   wpb  = (u16*)(ws + 67108864 + 12582912);              // 0.5 MB
  float* comb = (float*)(ws + 67108864 + 12582912 + 524288);   // 2 MB

  cvt_weights_kernel<<<6400, 256, 0, stream>>>(Wqkv, Wp, wqb, wpb,
                                               NEXP * QKVD * DIM, DIM * DIM);
  gate_convert_kernel<<<8192, 256, 0, stream>>>(x, Wg, bg, ebias, xbf, comb);
  moe_attn_proj_kernel<<<2048, 512, 0, stream>>>(xbf, wqb, wpb, comb, bp, out);
}
